// _CrossBlock_37580963840146
// MI455X (gfx1250) — matrix operand order repaired and checked by reading, not compiled, not run on hardware
//
#include <hip/hip_runtime.h>
#include <hip/hip_bf16.h>
#include <stdint.h>

#define D_MODEL 1024
#define NHEAD   16
#define HD      64
#define B_SZ    4
#define L_Q     512
#define T_KV    2048
#define D_FF    4096

typedef __attribute__((ext_vector_type(16))) __bf16 v16bf;
typedef __attribute__((ext_vector_type(8)))  float  v8f;
typedef unsigned short u16;
typedef unsigned int   u32;

__device__ __forceinline__ u16 f2bf(float f) {
  u32 u = __float_as_uint(f);
  u32 r = u + 0x7FFFu + ((u >> 16) & 1u);   // round-to-nearest-even
  return (u16)(r >> 16);
}

union Frag { uint4 q[2]; v16bf v; };

// A fragment: 16x32 (MxK) from row-major bf16, leading dim ld (elements).
// lane: row = lane&15; k-chunks {8h..8h+7} and {8h+16..8h+23}, h = lane>>4.
__device__ __forceinline__ v16bf load_a(const u16* p, int ld) {
  int lane = threadIdx.x & 31;
  const u16* s = p + (size_t)(lane & 15) * ld + (lane >> 4) * 8;
  Frag f;
  f.q[0] = *(const uint4*)(s);
  f.q[1] = *(const uint4*)(s + 16);
  return f.v;
}

// B fragment: 32x16 (KxN) where column n is row n of a row-major [N,K] source.
// lane: col = lane&15; k-chunks {8h..8h+7} and {8h+16..8h+23}, h = lane>>4.
__device__ __forceinline__ v16bf load_b(const u16* p, int ld) {
  int lane = threadIdx.x & 31;
  const u16* s = p + (size_t)(lane & 15) * ld + (lane >> 4) * 8;
  Frag f;
  f.q[0] = *(const uint4*)(s);
  f.q[1] = *(const uint4*)(s + 16);
  return f.v;
}

// LDS variant with scalar u16 loads (alias-safe vs the u16 stores that fill it).
__device__ __forceinline__ v16bf load_a_lds(const u16* p, int ld) {
  int lane = threadIdx.x & 31;
  const u16* s = p + (lane & 15) * ld + (lane >> 4) * 8;
  Frag f; u16* d = (u16*)&f;
#pragma unroll
  for (int i = 0; i < 8; ++i) d[i] = s[i];
#pragma unroll
  for (int i = 0; i < 8; ++i) d[8 + i] = s[16 + i];
  return f.v;
}

__device__ __forceinline__ v8f wmma_bf16(v16bf a, v16bf b, v8f c) {
  return __builtin_amdgcn_wmma_f32_16x16x32_bf16(false, a, false, b,
                                                 (short)0, c, false, false);
}

// ---------------------------------------------------------------- convert
__global__ __launch_bounds__(256) void cvt_f32_bf16(const float* __restrict__ src,
                                                    u16* __restrict__ dst, int n) {
  int stride = gridDim.x * blockDim.x;
  for (int i = blockIdx.x * blockDim.x + threadIdx.x; i < n; i += stride)
    dst[i] = f2bf(src[i]);
}

// ---------------------------------------------------------------- layernorm
__global__ __launch_bounds__(256) void ln_to_bf16(const float* __restrict__ x,
                                                  const float* __restrict__ g,
                                                  const float* __restrict__ b,
                                                  u16* __restrict__ out) {
  int row = blockIdx.x, tid = threadIdx.x;
  const float* xr = x + (size_t)row * D_MODEL;
  float4 v = ((const float4*)xr)[tid];          // 256 thr * 4 = 1024
  float s  = v.x + v.y + v.z + v.w;
  float s2 = v.x * v.x + v.y * v.y + v.z * v.z + v.w * v.w;
#pragma unroll
  for (int m = 1; m < 32; m <<= 1) { s += __shfl_xor(s, m, 32); s2 += __shfl_xor(s2, m, 32); }
  __shared__ float rs[8], rs2[8];
  int w = tid >> 5;
  if ((tid & 31) == 0) { rs[w] = s; rs2[w] = s2; }
  __syncthreads();
  if (w == 0) {
    float a  = (tid < 8) ? rs[tid]  : 0.f;
    float a2 = (tid < 8) ? rs2[tid] : 0.f;
#pragma unroll
    for (int m = 1; m < 8; m <<= 1) { a += __shfl_xor(a, m, 32); a2 += __shfl_xor(a2, m, 32); }
    if (tid == 0) { rs[0] = a; rs2[0] = a2; }
  }
  __syncthreads();
  float mean = rs[0] * (1.0f / D_MODEL);
  float var  = rs2[0] * (1.0f / D_MODEL) - mean * mean;
  float inv  = rsqrtf(var + 1e-5f);
  float4 gv = ((const float4*)g)[tid];
  float4 bv = ((const float4*)b)[tid];
  u16* o = out + (size_t)row * D_MODEL + tid * 4;
  o[0] = f2bf((v.x - mean) * inv * gv.x + bv.x);
  o[1] = f2bf((v.y - mean) * inv * gv.y + bv.y);
  o[2] = f2bf((v.z - mean) * inv * gv.z + bv.z);
  o[3] = f2bf((v.w - mean) * inv * gv.w + bv.w);
}

// ---------------------------------------------------------------- GEMM
// C[M,N] = A[M,K] @ W[N,K]^T + bias.  Block tile 128(M) x 128(N), 8 waves in a
// 4x2 grid; each wave a 32x64 strip (2x4 accumulators, 8 WMMAs / k-step).
// k-loop unrolled x2 with ping-pong fragment sets: no register rotation copies,
// loads for one stage in flight behind the other stage's WMMAs.
// Requires K % 64 == 0 (true for all call sites: 1024, 4096).
// mode 0: bf16 out      mode 1: fp32 out + residual
// mode 2: GELU -> bf16  mode 3: bf16 transposed per-head (V^T layout)
__global__ __launch_bounds__(256) void gemm_bf16_wmma(
    const u16* __restrict__ A, int lda, const u16* __restrict__ W, int ldw,
    const float* __restrict__ bias, int M, int N, int K,
    float* __restrict__ outF, u16* __restrict__ outB,
    const float* __restrict__ res, int mode, int Tdim) {
  int w = threadIdx.x >> 5;
  int lane = threadIdx.x & 31;
  int r = lane & 15, half = lane >> 4;
  int m0 = blockIdx.y * 128 + (w >> 1) * 32;
  int n0 = blockIdx.x * 128 + (w & 1) * 64;

  const u16* Ap0 = A + (size_t)m0 * lda;
  const u16* Ap1 = Ap0 + (size_t)16 * lda;
  const u16* Wp  = W + (size_t)n0 * ldw;

  v8f acc[2][4] = {};
  // stage A fragments (k = 0)
  v16bf a0 = load_a(Ap0, lda), a1 = load_a(Ap1, lda);
  v16bf bA[4];
#pragma unroll
  for (int nt = 0; nt < 4; ++nt) bA[nt] = load_b(Wp + (size_t)nt * 16 * ldw, ldw);
  v16bf c0, c1, bB[4];

  int k0 = 0;
  for (; k0 < K - 64; k0 += 64) {
    // stage B loads (k0+32) issue, then stage A WMMAs run under them
    c0 = load_a(Ap0 + k0 + 32, lda);
    c1 = load_a(Ap1 + k0 + 32, lda);
#pragma unroll
    for (int nt = 0; nt < 4; ++nt)
      bB[nt] = load_b(Wp + (size_t)nt * 16 * ldw + k0 + 32, ldw);
#pragma unroll
    for (int nt = 0; nt < 4; ++nt) {
      acc[0][nt] = wmma_bf16(a0, bA[nt], acc[0][nt]);
      acc[1][nt] = wmma_bf16(a1, bA[nt], acc[1][nt]);
    }
    // stage A loads (k0+64) issue, then stage B WMMAs run under them
    a0 = load_a(Ap0 + k0 + 64, lda);
    a1 = load_a(Ap1 + k0 + 64, lda);
#pragma unroll
    for (int nt = 0; nt < 4; ++nt)
      bA[nt] = load_b(Wp + (size_t)nt * 16 * ldw + k0 + 64, ldw);
#pragma unroll
    for (int nt = 0; nt < 4; ++nt) {
      acc[0][nt] = wmma_bf16(c0, bB[nt], acc[0][nt]);
      acc[1][nt] = wmma_bf16(c1, bB[nt], acc[1][nt]);
    }
  }
  // tail: last 64 of K (stage A already holds k0; stage B = k0+32)
  c0 = load_a(Ap0 + k0 + 32, lda);
  c1 = load_a(Ap1 + k0 + 32, lda);
#pragma unroll
  for (int nt = 0; nt < 4; ++nt)
    bB[nt] = load_b(Wp + (size_t)nt * 16 * ldw + k0 + 32, ldw);
#pragma unroll
  for (int nt = 0; nt < 4; ++nt) {
    acc[0][nt] = wmma_bf16(a0, bA[nt], acc[0][nt]);
    acc[1][nt] = wmma_bf16(a1, bA[nt], acc[1][nt]);
  }
#pragma unroll
  for (int nt = 0; nt < 4; ++nt) {
    acc[0][nt] = wmma_bf16(c0, bB[nt], acc[0][nt]);
    acc[1][nt] = wmma_bf16(c1, bB[nt], acc[1][nt]);
  }

#pragma unroll
  for (int mi = 0; mi < 2; ++mi) {
#pragma unroll
    for (int nt = 0; nt < 4; ++nt) {
      int n = n0 + nt * 16 + r;
      float bv = bias[n];
#pragma unroll
      for (int j = 0; j < 8; ++j) {
        int m = m0 + mi * 16 + j + 8 * half;
        float v = acc[mi][nt][j] + bv;
        size_t idx = (size_t)m * N + n;
        if (mode == 0) {
          outB[idx] = f2bf(v);
        } else if (mode == 1) {
          outF[idx] = v + res[idx];
        } else if (mode == 2) {
          outB[idx] = f2bf(0.5f * v * (1.0f + erff(v * 0.70710678118654752f)));
        } else {                       // mode 3: vT[b,h,d,t]
          int b_ = m / Tdim, t = m - b_ * Tdim;
          int h_ = n >> 6,  d = n & 63;
          outB[((size_t)(b_ * NHEAD + h_) * HD + d) * (size_t)Tdim + t] = f2bf(v);
        }
      }
    }
  }
}

// ---------------------------------------------------------------- attention
// grid (L/128, H, B), 8 waves; each wave owns a 16-row L tile, flash loop over T.
// K-fragments are pipelined one t-step ahead; V-fragments load before the
// softmax VALU chain so both streams hide behind compute.
__global__ __launch_bounds__(256) void attn_wmma(
    const u16* __restrict__ qp, const u16* __restrict__ kp,
    const u16* __restrict__ vT, const float* __restrict__ qc,
    const float* __restrict__ tc, const unsigned char* __restrict__ mask,
    const float* __restrict__ sscale, const float* __restrict__ soff,
    u16* __restrict__ attn) {
  int b = blockIdx.z, h = blockIdx.y;
  int w = threadIdx.x >> 5;
  int lane = threadIdx.x & 31;
  int r = lane & 15, half = lane >> 4;
  int l0 = blockIdx.x * 128 + w * 16;

  __shared__ __align__(16) u16 pbuf[8][16 * 32];
  u16* pb = pbuf[w];

  const float scl = sscale[h], off = soff[h];
  const float inv_sqrt = 0.125f;          // 1/sqrt(64)

  const u16* qbase = qp + ((size_t)(b * L_Q + l0)) * D_MODEL + h * HD;
  v16bf qa0 = load_a(qbase, D_MODEL);
  v16bf qa1 = load_a(qbase + 32, D_MODEL);

  float qx[8], qy[8], mi[8], si[8];
#pragma unroll
  for (int j = 0; j < 8; ++j) {
    int m = l0 + j + 8 * half;
    qx[j] = qc[((size_t)b * L_Q + m) * 2 + 0];
    qy[j] = qc[((size_t)b * L_Q + m) * 2 + 1];
    mi[j] = -1e30f; si[j] = 0.f;
  }
  v8f o0 = {}, o1 = {}, o2 = {}, o3 = {};
  const u16* kh = kp + (size_t)b * T_KV * D_MODEL + h * HD;
  const u16* vh = vT + (size_t)(b * NHEAD + h) * HD * T_KV;

  // prologue: K fragments for t0 = 0
  v16bf kf0 = load_b(kh, D_MODEL);
  v16bf kf1 = load_b(kh + 32, D_MODEL);
  v16bf kf2 = load_b(kh + (size_t)16 * D_MODEL, D_MODEL);
  v16bf kf3 = load_b(kh + (size_t)16 * D_MODEL + 32, D_MODEL);

  for (int t0 = 0; t0 < T_KV; t0 += 32) {
    v8f z = {};
    v8f s0 = wmma_bf16(qa0, kf0, z);
    s0     = wmma_bf16(qa1, kf1, s0);
    v8f s1 = wmma_bf16(qa0, kf2, z);
    s1     = wmma_bf16(qa1, kf3, s1);

    // V fragments for this step + K fragments for the next step: in flight
    // while the softmax VALU chain below executes.
    v16bf vb0 = load_b(vh + (size_t)0  * T_KV + t0, T_KV);
    v16bf vb1 = load_b(vh + (size_t)16 * T_KV + t0, T_KV);
    v16bf vb2 = load_b(vh + (size_t)32 * T_KV + t0, T_KV);
    v16bf vb3 = load_b(vh + (size_t)48 * T_KV + t0, T_KV);
    {
      int tn = (t0 + 32 < T_KV) ? t0 + 32 : t0;     // clamp: dummy reload on last
      const u16* kb0 = kh + (size_t)tn * D_MODEL;
      const u16* kb1 = kb0 + (size_t)16 * D_MODEL;
      kf0 = load_b(kb0, D_MODEL);
      kf1 = load_b(kb0 + 32, D_MODEL);
      kf2 = load_b(kb1, D_MODEL);
      kf3 = load_b(kb1 + 32, D_MODEL);
    }

    int tA = t0 + r, tB = t0 + 16 + r;
    float txA = tc[((size_t)b * T_KV + tA) * 2], tyA = tc[((size_t)b * T_KV + tA) * 2 + 1];
    float txB = tc[((size_t)b * T_KV + tB) * 2], tyB = tc[((size_t)b * T_KV + tB) * 2 + 1];
    bool mkA = mask[(size_t)b * T_KV + tA] != 0;
    bool mkB = mask[(size_t)b * T_KV + tB] != 0;

#pragma unroll
    for (int j = 0; j < 8; ++j) {
      float dxA = qx[j] - txA, dyA = qy[j] - tyA;
      float dxB = qx[j] - txB, dyB = qy[j] - tyB;
      float vA = s0[j] * inv_sqrt + off - scl * sqrtf(dxA * dxA + dyA * dyA);
      float vB = s1[j] * inv_sqrt + off - scl * sqrtf(dxB * dxB + dyB * dyB);
      if (!mkA) vA = -1e30f;
      if (!mkB) vB = -1e30f;
      float rm = fmaxf(vA, vB);
#pragma unroll
      for (int mk = 1; mk < 16; mk <<= 1) rm = fmaxf(rm, __shfl_xor(rm, mk, 32));
      float mn = fmaxf(mi[j], rm);
      float sc = __expf(mi[j] - mn);
      float pA = __expf(vA - mn), pB = __expf(vB - mn);
      float rsum = pA + pB;
#pragma unroll
      for (int mk = 1; mk < 16; mk <<= 1) rsum += __shfl_xor(rsum, mk, 32);
      si[j] = si[j] * sc + rsum;
      mi[j] = mn;
      o0[j] *= sc; o1[j] *= sc; o2[j] *= sc; o3[j] *= sc;
      int row = j + 8 * half;
      pb[row * 32 + r]      = f2bf(pA);
      pb[row * 32 + 16 + r] = f2bf(pB);
    }
    v16bf pa = load_a_lds(pb, 32);                    // same-wave LDS RAW: in-order
    o0 = wmma_bf16(pa, vb0, o0);
    o1 = wmma_bf16(pa, vb1, o1);
    o2 = wmma_bf16(pa, vb2, o2);
    o3 = wmma_bf16(pa, vb3, o3);
  }
#pragma unroll
  for (int j = 0; j < 8; ++j) {
    float inv = 1.0f / si[j];
    int m = l0 + j + 8 * half;
    u16* orow = attn + ((size_t)(b * L_Q + m)) * D_MODEL + h * HD;
    orow[r]      = f2bf(o0[j] * inv);
    orow[16 + r] = f2bf(o1[j] * inv);
    orow[32 + r] = f2bf(o2[j] * inv);
    orow[48 + r] = f2bf(o3[j] * inv);
  }
}

// ---------------------------------------------------------------- launcher
extern "C" void kernel_launch(void* const* d_in, const int* in_sizes, int n_in,
                              void* d_out, int out_size, void* d_ws, size_t ws_size,
                              hipStream_t stream) {
  const float* queries = (const float*)d_in[0];
  const float* tokens  = (const float*)d_in[1];
  const unsigned char* token_mask = (const unsigned char*)d_in[2];
  const float* qc   = (const float*)d_in[3];
  const float* tc   = (const float*)d_in[4];
  const float* qng  = (const float*)d_in[5];
  const float* qnb  = (const float*)d_in[6];
  const float* kvg  = (const float*)d_in[7];
  const float* kvb  = (const float*)d_in[8];
  const float* in_w = (const float*)d_in[9];
  const float* in_b = (const float*)d_in[10];
  const float* ow   = (const float*)d_in[11];
  const float* ob   = (const float*)d_in[12];
  const float* sscale = (const float*)d_in[13];
  const float* soff   = (const float*)d_in[14];
  const float* mng  = (const float*)d_in[15];
  const float* mnb  = (const float*)d_in[16];
  const float* w1   = (const float*)d_in[17];
  const float* b1   = (const float*)d_in[18];
  const float* w2   = (const float*)d_in[19];
  const float* b2   = (const float*)d_in[20];

  char* ws = (char*)d_ws;
  size_t off = 0;
  auto alloc = [&](size_t bytes) -> void* {
    void* p = ws + off; off += (bytes + 255) & ~(size_t)255; return p;
  };
  const int ML = B_SZ * L_Q;     // 2048
  const int MT = B_SZ * T_KV;    // 8192
  u16* Wqkv = (u16*)alloc((size_t)3 * D_MODEL * D_MODEL * 2);
  u16* Wout = (u16*)alloc((size_t)D_MODEL * D_MODEL * 2);
  u16* W1b  = (u16*)alloc((size_t)D_FF * D_MODEL * 2);
  u16* W2b  = (u16*)alloc((size_t)D_MODEL * D_FF * 2);
  u16* qln  = (u16*)alloc((size_t)ML * D_MODEL * 2);
  u16* kvln = (u16*)alloc((size_t)MT * D_MODEL * 2);
  u16* qpb  = (u16*)alloc((size_t)ML * D_MODEL * 2);
  u16* kpb  = (u16*)alloc((size_t)MT * D_MODEL * 2);
  u16* vTb  = (u16*)alloc((size_t)MT * D_MODEL * 2);
  u16* attn = (u16*)alloc((size_t)ML * D_MODEL * 2);
  float* xbuf = (float*)alloc((size_t)ML * D_MODEL * 4);
  u16* xln  = (u16*)alloc((size_t)ML * D_MODEL * 2);
  u16* h1b  = (u16*)alloc((size_t)ML * D_FF * 2);

  // 1. weight conversion
  cvt_f32_bf16<<<2048, 256, 0, stream>>>(in_w, Wqkv, 3 * D_MODEL * D_MODEL);
  cvt_f32_bf16<<<1024, 256, 0, stream>>>(ow, Wout, D_MODEL * D_MODEL);
  cvt_f32_bf16<<<2048, 256, 0, stream>>>(w1, W1b, D_FF * D_MODEL);
  cvt_f32_bf16<<<2048, 256, 0, stream>>>(w2, W2b, D_MODEL * D_FF);
  // 2. input layernorms
  ln_to_bf16<<<ML, 256, 0, stream>>>(queries, qng, qnb, qln);
  ln_to_bf16<<<MT, 256, 0, stream>>>(tokens, kvg, kvb, kvln);
  // 3. Q/K/V projections
  gemm_bf16_wmma<<<dim3(D_MODEL / 128, ML / 128), 256, 0, stream>>>(
      qln, D_MODEL, Wqkv, D_MODEL, in_b, ML, D_MODEL, D_MODEL,
      nullptr, qpb, nullptr, 0, 0);
  gemm_bf16_wmma<<<dim3(D_MODEL / 128, MT / 128), 256, 0, stream>>>(
      kvln, D_MODEL, Wqkv + (size_t)D_MODEL * D_MODEL, D_MODEL, in_b + D_MODEL,
      MT, D_MODEL, D_MODEL, nullptr, kpb, nullptr, 0, 0);
  gemm_bf16_wmma<<<dim3(D_MODEL / 128, MT / 128), 256, 0, stream>>>(
      kvln, D_MODEL, Wqkv + (size_t)2 * D_MODEL * D_MODEL, D_MODEL,
      in_b + 2 * D_MODEL, MT, D_MODEL, D_MODEL, nullptr, vTb, nullptr, 3, T_KV);
  // 4. flash attention
  attn_wmma<<<dim3(L_Q / 128, NHEAD, B_SZ), 256, 0, stream>>>(
      qpb, kpb, vTb, qc, tc, token_mask, sscale, soff, attn);
  // 5. out_proj + residual
  gemm_bf16_wmma<<<dim3(D_MODEL / 128, ML / 128), 256, 0, stream>>>(
      attn, D_MODEL, Wout, D_MODEL, ob, ML, D_MODEL, D_MODEL,
      xbuf, nullptr, queries, 1, 0);
  // 6. MLP
  ln_to_bf16<<<ML, 256, 0, stream>>>(xbuf, mng, mnb, xln);
  gemm_bf16_wmma<<<dim3(D_FF / 128, ML / 128), 256, 0, stream>>>(
      xln, D_MODEL, W1b, D_MODEL, b1, ML, D_FF, D_MODEL,
      nullptr, h1b, nullptr, 2, 0);
  gemm_bf16_wmma<<<dim3(D_MODEL / 128, ML / 128), 256, 0, stream>>>(
      h1b, D_FF, W2b, D_FF, b2, ML, D_MODEL, D_FF,
      (float*)d_out, nullptr, xbuf, 1, 0);
}
